// VideoMambaBlock_78950088835744
// MI455X (gfx1250) — matrix operand order repaired and checked by reading, not compiled, not run on hardware
//
#include <hip/hip_runtime.h>
#include <hip/hip_bf16.h>

typedef __attribute__((ext_vector_type(16))) _Float16 v16h;
typedef __attribute__((ext_vector_type(8)))  _Float16 v8h;
typedef __attribute__((ext_vector_type(8)))  float    v8f;
typedef __attribute__((ext_vector_type(2)))  float    v2f;

#define DM   256      // D_MODEL
#define DS   16       // D_STATE
#define DI   512      // D_INNER
#define DR   16       // DT_RANK
#define NX   48       // DT_RANK + 2*D_STATE
#define N1   1024     // 2*D_INNER
#define BATCH 8
#define SEQL 4096
#define BL   (BATCH*SEQL)
#define BPAD 40       // padded LDS row stride (halves): 20 dwords -> conflict-free

// ---------------------------------------------------------------- helpers

__device__ __forceinline__ float silu_f(float x) {
    return x / (1.0f + __expf(-x));
}

// A-fragment: 16x32 f16, row-major source with stride `ld` (halves).
// lane m=lane&15 holds K = kb..kb+7 and kb+16..kb+23, kb=(lane>>4)*8
__device__ __forceinline__ v16h load_fragA(const _Float16* __restrict__ base,
                                           int ld, int lane, int k) {
    int r  = lane & 15;
    int kb = (lane >> 4) * 8;
    const _Float16* p = base + (size_t)r * ld + k + kb;
    v8h lo = *(const v8h*)(p);
    v8h hi = *(const v8h*)(p + 16);
    v16h f;
#pragma unroll
    for (int i = 0; i < 8; ++i) { f[i] = lo[i]; f[i + 8] = hi[i]; }
    return f;
}

// B-fragment from an LDS-staged tile (row stride BPAD halves).
// lane n=lane&15 = column, holds K = kb..kb+7 and kb+16..kb+23, kb=(lane>>4)*8
__device__ __forceinline__ v16h load_fragB_lds(const _Float16* wbase, int lane) {
    int n  = lane & 15;
    int kb = (lane >> 4) * 8;
    const _Float16* p = wbase + n * BPAD + kb;
    v8h lo = *(const v8h*)(p);
    v8h hi = *(const v8h*)(p + 16);
    v16h f;
#pragma unroll
    for (int i = 0; i < 8; ++i) { f[i] = lo[i]; f[i + 8] = hi[i]; }
    return f;
}

// ---------------------------------------------------------------- kernels

// f32 -> f16 conversion (weights)
__global__ void k_cvt_f16(const float* __restrict__ src, _Float16* __restrict__ dst, int n) {
    int i = blockIdx.x * blockDim.x + threadIdx.x;
    if (i < n) dst[i] = (_Float16)src[i];
}

// x (b, C=256, P=4096) -> xs_f16 (b*P, 256)
__global__ void k_transpose_x(const float* __restrict__ x, _Float16* __restrict__ xs) {
    int idx = blockIdx.x * blockDim.x + threadIdx.x;     // p fastest -> coalesced read
    if (idx >= BATCH * DM * SEQL) return;
    int p = idx & (SEQL - 1);
    int t = idx >> 12;
    int c = t & (DM - 1);
    int b = t >> 8;
    xs[((size_t)(b * SEQL + p)) * DM + c] = (_Float16)x[idx];
}

// GEMM1: xz(BL,1024) = xs(BL,256) @ W_in(1024,256)^T   [f16 WMMA, f32 acc]
// Block tile 128(M)x64(N); B tile staged in LDS, shared by all 8 waves.
__global__ void k_gemm_xz(const _Float16* __restrict__ A, const _Float16* __restrict__ W,
                          float* __restrict__ C) {
    __shared__ _Float16 Bs[64 * BPAD];
    int tid = threadIdx.x;
    int lane = tid & 31, wave = tid >> 5;
    int m0 = blockIdx.y * 128 + wave * 16;
    int n0 = blockIdx.x * 64;
    int brow = tid >> 2;            // 0..63
    int bcol = (tid & 3) * 8;       // 0,8,16,24
    v8f acc[4] = {};
    const _Float16* Ab = A + (size_t)m0 * DM;
#pragma unroll
    for (int k = 0; k < DM; k += 32) {
        *(v8h*)&Bs[brow * BPAD + bcol] =
            *(const v8h*)(W + (size_t)(n0 + brow) * DM + k + bcol);
        __syncthreads();
        v16h a = load_fragA(Ab, DM, lane, k);
#pragma unroll
        for (int j = 0; j < 4; ++j) {
            v16h b = load_fragB_lds(&Bs[j * 16 * BPAD], lane);
            acc[j] = __builtin_amdgcn_wmma_f32_16x16x32_f16(
                false, a, false, b, (short)0, acc[j], false, false);
        }
        __syncthreads();
    }
    int mlo = (lane >> 4) * 8, n = lane & 15;
#pragma unroll
    for (int j = 0; j < 4; ++j)
#pragma unroll
        for (int r = 0; r < 8; ++r)
            C[(size_t)(m0 + mlo + r) * N1 + n0 + j * 16 + n] = acc[j][r];
}

// depthwise causal conv(k=4) + bias + SiLU on u-half of xz
__global__ void k_conv_silu(const float* __restrict__ xz, const float* __restrict__ cw,
                            const float* __restrict__ cb, float* __restrict__ u_f,
                            _Float16* __restrict__ u_h) {
    int idx = blockIdx.x * blockDim.x + threadIdx.x;     // over BL*512, d fastest
    if (idx >= BL * DI) return;
    int d  = idx & (DI - 1);
    int bl = idx >> 9;
    int p  = bl & (SEQL - 1);
    float acc = cb[d];
#pragma unroll
    for (int j = 0; j < 4; ++j) {
        int pp = p - 3 + j;
        if (pp >= 0)
            acc = fmaf(cw[d * 4 + j], xz[((size_t)(bl - 3 + j)) * N1 + d], acc);
    }
    float s = silu_f(acc);
    u_f[idx] = s;
    u_h[idx] = (_Float16)s;
}

// x_dbl(BL,48) = u(BL,512) @ W_xproj(48,512)^T   [f16 WMMA, LDS-staged B]
__global__ void k_gemm_xdbl(const _Float16* __restrict__ A, const _Float16* __restrict__ W,
                            float* __restrict__ C) {
    __shared__ _Float16 Bs[48 * BPAD];
    int tid = threadIdx.x;
    int lane = tid & 31, wave = tid >> 5;
    int m0 = blockIdx.x * 128 + wave * 16;
    int brow = tid >> 2;            // 0..63 (only 0..47 used)
    int bcol = (tid & 3) * 8;
    v8f acc[3] = {};
    const _Float16* Ab = A + (size_t)m0 * DI;
    for (int k = 0; k < DI; k += 32) {
        if (brow < NX)
            *(v8h*)&Bs[brow * BPAD + bcol] =
                *(const v8h*)(W + (size_t)brow * DI + k + bcol);
        __syncthreads();
        v16h a = load_fragA(Ab, DI, lane, k);
#pragma unroll
        for (int j = 0; j < 3; ++j) {
            v16h b = load_fragB_lds(&Bs[j * 16 * BPAD], lane);
            acc[j] = __builtin_amdgcn_wmma_f32_16x16x32_f16(
                false, a, false, b, (short)0, acc[j], false, false);
        }
        __syncthreads();
    }
    int mlo = (lane >> 4) * 8, n = lane & 15;
#pragma unroll
    for (int j = 0; j < 3; ++j)
#pragma unroll
        for (int r = 0; r < 8; ++r)
            C[(size_t)(m0 + mlo + r) * NX + j * 16 + n] = acc[j][r];
}

// dt(BL,512) = softplus(x_dbl[:, :16](BL,16) @ W_dt(512,16)^T + b_dt)  [f32 WMMA 16x16x4]
__global__ void k_gemm_dt(const float* __restrict__ xdbl, const float* __restrict__ Wdt,
                          const float* __restrict__ bdt, float* __restrict__ dt) {
    int lane = threadIdx.x & 31, wave = threadIdx.x >> 5;
    int m0 = blockIdx.y * 128 + wave * 16;
    int n0 = blockIdx.x * 64;
    int r  = lane & 15;
    int k0 = (lane >> 4) * 2;
    v8f acc[4] = {};
#pragma unroll
    for (int k = 0; k < DR; k += 4) {
        v2f a;
        a[0] = xdbl[(size_t)(m0 + r) * NX + k + k0];
        a[1] = xdbl[(size_t)(m0 + r) * NX + k + k0 + 1];
#pragma unroll
        for (int j = 0; j < 4; ++j) {
            int nc = n0 + j * 16 + r;     // r == lane&15 doubles as B column
            v2f b;
            b[0] = Wdt[(size_t)nc * DR + k + k0];
            b[1] = Wdt[(size_t)nc * DR + k + k0 + 1];
            acc[j] = __builtin_amdgcn_wmma_f32_16x16x4_f32(
                false, a, false, b, (short)0, acc[j], false, false);
        }
    }
    int mlo = (lane >> 4) * 8, n = lane & 15;
#pragma unroll
    for (int j = 0; j < 4; ++j) {
        int col = n0 + j * 16 + n;
        float bb = bdt[col];
#pragma unroll
        for (int rr = 0; rr < 8; ++rr) {
            float v = acc[j][rr] + bb;
            float sp = (v > 20.f) ? v : log1pf(__expf(v));   // softplus
            dt[(size_t)(m0 + mlo + rr) * DI + col] = sp;
        }
    }
}

// selective scan: one block per (b, d-half of 256). Sequential over L.
__global__ void k_scan(const float* __restrict__ xdbl, const float* __restrict__ dt_f,
                       const float* __restrict__ u_f, const float* __restrict__ xz_f,
                       const float* __restrict__ A_log, const float* __restrict__ Dp,
                       _Float16* __restrict__ y_h) {
    __shared__ float Bs[DS];
    __shared__ float Cs[DS];
    int b = blockIdx.x >> 1;
    int d = (blockIdx.x & 1) * 256 + threadIdx.x;
    float Ad[DS];
#pragma unroll
    for (int n = 0; n < DS; ++n) Ad[n] = -__expf(A_log[d * DS + n]);
    float Dd = Dp[d];
    float h[DS];
#pragma unroll
    for (int n = 0; n < DS; ++n) h[n] = 0.f;

    const float* xdb = xdbl + (size_t)b * SEQL * NX;
    for (int t = 0; t < SEQL; ++t) {
        if (threadIdx.x < 32) {
            int l = threadIdx.x;
            float v = xdb[(size_t)t * NX + DR + l];   // cols 16..47 = B then C
            if (l < DS) Bs[l] = v; else Cs[l - DS] = v;
        }
        __syncthreads();
        size_t row = (size_t)b * SEQL + t;
        float dt = dt_f[row * DI + d];
        float uv = u_f[row * DI + d];
        if (t + 4 < SEQL) {   // software prefetch of future timesteps -> global_prefetch_b8
            __builtin_prefetch(&dt_f[(row + 4) * DI + d], 0, 1);
            __builtin_prefetch(&u_f[(row + 4) * DI + d], 0, 1);
            __builtin_prefetch(&xz_f[(row + 4) * N1 + DI + d], 0, 1);
        }
        float du = dt * uv;
        float y = 0.f;
#pragma unroll
        for (int n = 0; n < DS; ++n) {
            h[n] = fmaf(h[n], __expf(dt * Ad[n]), du * Bs[n]);
            y = fmaf(h[n], Cs[n], y);
        }
        float z = xz_f[row * N1 + DI + d];
        y = (y + uv * Dd) * silu_f(z);
        y_h[row * DI + d] = (_Float16)y;
        __syncthreads();
    }
}

// out = y(BL,512) @ W_out(256,512)^T, stored transposed into (b, 256, 4096)
__global__ void k_gemm_out(const _Float16* __restrict__ A, const _Float16* __restrict__ W,
                           float* __restrict__ out) {
    __shared__ _Float16 Bs[64 * BPAD];
    int tid = threadIdx.x;
    int lane = tid & 31, wave = tid >> 5;
    int m0 = blockIdx.y * 128 + wave * 16;
    int n0 = blockIdx.x * 64;
    int brow = tid >> 2;
    int bcol = (tid & 3) * 8;
    v8f acc[4] = {};
    const _Float16* Ab = A + (size_t)m0 * DI;
    for (int k = 0; k < DI; k += 32) {
        *(v8h*)&Bs[brow * BPAD + bcol] =
            *(const v8h*)(W + (size_t)(n0 + brow) * DI + k + bcol);
        __syncthreads();
        v16h a = load_fragA(Ab, DI, lane, k);
#pragma unroll
        for (int j = 0; j < 4; ++j) {
            v16h b = load_fragB_lds(&Bs[j * 16 * BPAD], lane);
            acc[j] = __builtin_amdgcn_wmma_f32_16x16x32_f16(
                false, a, false, b, (short)0, acc[j], false, false);
        }
        __syncthreads();
    }
    int mlo = (lane >> 4) * 8, n = lane & 15;
#pragma unroll
    for (int j = 0; j < 4; ++j)
#pragma unroll
        for (int r = 0; r < 8; ++r) {
            int m = m0 + mlo + r;
            int b = m >> 12, p = m & (SEQL - 1);
            int c = n0 + j * 16 + n;
            out[((size_t)b * DM + c) * SEQL + p] = acc[j][r];
        }
}

// ---------------------------------------------------------------- launch

extern "C" void kernel_launch(void* const* d_in, const int* in_sizes, int n_in,
                              void* d_out, int out_size, void* d_ws, size_t ws_size,
                              hipStream_t stream) {
    const float* x      = (const float*)d_in[0];
    const float* W_in   = (const float*)d_in[1];
    const float* conv_w = (const float*)d_in[2];
    const float* conv_b = (const float*)d_in[3];
    const float* W_xprj = (const float*)d_in[4];
    const float* W_dt   = (const float*)d_in[5];
    const float* b_dt   = (const float*)d_in[6];
    const float* A_log  = (const float*)d_in[7];
    const float* Dvec   = (const float*)d_in[8];
    const float* W_out  = (const float*)d_in[9];
    float* out = (float*)d_out;

    char* w = (char*)d_ws;
    size_t o = 0;
    auto alloc = [&](size_t bytes) -> char* {
        o = (o + 255) & ~(size_t)255;
        char* p = w + o;
        o += bytes;
        return p;
    };
    _Float16* Win_h  = (_Float16*)alloc((size_t)N1 * DM * 2);
    _Float16* Wxp_h  = (_Float16*)alloc((size_t)NX * DI * 2);
    _Float16* Wout_h = (_Float16*)alloc((size_t)DM * DI * 2);
    _Float16* xs_h   = (_Float16*)alloc((size_t)BL * DM * 2);
    float*    xz_f   = (float*)   alloc((size_t)BL * N1 * 4);
    float*    u_f    = (float*)   alloc((size_t)BL * DI * 4);
    _Float16* u_h    = (_Float16*)alloc((size_t)BL * DI * 2);
    float*    xdbl_f = (float*)   alloc((size_t)BL * NX * 4);
    float*    dt_f   = (float*)   alloc((size_t)BL * DI * 4);
    _Float16* y_h    = (_Float16*)alloc((size_t)BL * DI * 2);
    (void)ws_size; (void)in_sizes; (void)n_in; (void)out_size;

    // weight conversions
    k_cvt_f16<<<(N1 * DM + 255) / 256, 256, 0, stream>>>(W_in, Win_h, N1 * DM);
    k_cvt_f16<<<(NX * DI + 255) / 256, 256, 0, stream>>>(W_xprj, Wxp_h, NX * DI);
    k_cvt_f16<<<(DM * DI + 255) / 256, 256, 0, stream>>>(W_out, Wout_h, DM * DI);

    // (b,c,p) -> (bl, c) f16
    k_transpose_x<<<(BATCH * DM * SEQL) / 256, 256, 0, stream>>>(x, xs_h);

    // xz = xs @ W_in^T
    k_gemm_xz<<<dim3(N1 / 64, BL / 128), 256, 0, stream>>>(xs_h, Win_h, xz_f);

    // depthwise conv + SiLU
    k_conv_silu<<<(BL * DI) / 256, 256, 0, stream>>>(xz_f, conv_w, conv_b, u_f, u_h);

    // x_dbl = u @ W_xproj^T
    k_gemm_xdbl<<<BL / 128, 256, 0, stream>>>(u_h, Wxp_h, xdbl_f);

    // dt = softplus(x_dbl[:, :16] @ W_dt^T + b_dt)
    k_gemm_dt<<<dim3(DI / 64, BL / 128), 256, 0, stream>>>(xdbl_f, W_dt, b_dt, dt_f);

    // selective scan + D skip + z gate -> y (f16)
    k_scan<<<BATCH * 2, 256, 0, stream>>>(xdbl_f, dt_f, u_f, xz_f, A_log, Dvec, y_h);

    // out = y @ W_out^T, transposed store
    k_gemm_out<<<dim3(DM / 64, BL / 128), 256, 0, stream>>>(y_h, Wout_h, out);
}
